// NonLocalBlock_21199958573716
// MI455X (gfx1250) — matrix operand order repaired and checked by reading, not compiled, not run on hardware
//
#include <hip/hip_runtime.h>
#include <hip/hip_bf16.h>

// ---------------------------------------------------------------------------
// Non-local block (self-attention), MI455X / gfx1250, wave32 + WMMA bf16.
// B=4, C=256, CI=128, N=4096. Flash-style fused softmax: never materializes
// the 256MB [B,N,N] score tensor (wouldn't fit 192MB L2; would be HBM-bound).
// All GEMMs use v_wmma_f32_16x16x32_bf16 with f32 accumulation.
// Attention K/V slabs are double-buffered into LDS with the CDNA5 async copy
// path (global_load_async_to_lds_b128 + s_wait_asynccnt). Weights are
// converted to bf16 once (k_wcvt) so GEMM B-fragments are pure b128 loads.
// ---------------------------------------------------------------------------

typedef __attribute__((ext_vector_type(16))) __bf16 v16bf;
typedef __attribute__((ext_vector_type(8)))  float  v8f;

#define WMMA_BF16(A, Bm, Cm) \
  __builtin_amdgcn_wmma_f32_16x16x32_bf16(false, (A), false, (Bm), (short)0, (Cm), false, false)

static constexpr int kB  = 4;
static constexpr int kC  = 256;
static constexpr int kCI = 128;
static constexpr int kN  = 4096;

// ---------------- WMMA fragment loaders (ISA 7.12.2 layouts, wave32) -------

// A fragment, 16x32 16-bit. elem(m,k) = src[m*stride + k]
static __device__ __forceinline__ v16bf load_a(const __bf16* src, int stride, int lane) {
  int m  = lane & 15;
  int hk = (lane >> 4) & 1;            // lanes 16-31 hold K blocks 8..15 / 24..31
  v16bf a;
#pragma unroll
  for (int v = 0; v < 8; ++v) {
    int k0 = ((v & 4) ? 16 : 0) + hk * 8 + 2 * (v & 3);
    a[2 * v]     = src[m * stride + k0];
    a[2 * v + 1] = src[m * stride + k0 + 1];
  }
  return a;
}

// A fragment from K-major source. elem(m,k) = src[k*stride + m]
static __device__ __forceinline__ v16bf load_aT(const __bf16* src, int stride, int lane) {
  int m  = lane & 15;
  int hk = (lane >> 4) & 1;
  v16bf a;
#pragma unroll
  for (int v = 0; v < 8; ++v) {
    int k0 = ((v & 4) ? 16 : 0) + hk * 8 + 2 * (v & 3);
    a[2 * v]     = src[(k0)     * stride + m];
    a[2 * v + 1] = src[(k0 + 1) * stride + m];
  }
  return a;
}

// B fragment from N-major source (K contiguous per lane -> b128 load pairs).
// elem(k,n) = src[n*stride + k]
static __device__ __forceinline__ v16bf load_bT(const __bf16* src, int stride, int lane) {
  int n  = lane & 15;
  int kb = ((lane >> 4) & 1) * 8;
  v16bf b;
#pragma unroll
  for (int v = 0; v < 8; ++v) {
    int k0 = ((v & 4) ? 16 : 0) + kb + 2 * (v & 3);
    b[2 * v]     = src[n * stride + k0];
    b[2 * v + 1] = src[n * stride + k0 + 1];
  }
  return b;
}

static __device__ __forceinline__ v8f vsplat8(float f) {
  v8f r;
#pragma unroll
  for (int e = 0; e < 8; ++e) r[e] = f;
  return r;
}

// Row reductions inside one 16-lane half (wave32; xor masks <=8 stay in half).
static __device__ __forceinline__ float half_max(float v) {
#pragma unroll
  for (int m = 1; m <= 8; m <<= 1) v = fmaxf(v, __shfl_xor(v, m, 32));
  return v;
}
static __device__ __forceinline__ float half_sum(float v) {
#pragma unroll
  for (int m = 1; m <= 8; m <<= 1) v += __shfl_xor(v, m, 32);
  return v;
}

// CDNA5 async copy: 16B global -> LDS, tracked by ASYNCcnt (bypasses VGPRs).
// GVS addressing: mem = SADDR(sgpr64) + VADDR(u32, bytes) + IOFFSET;
// LDS dest = VDST(u32 lds byte addr) + IOFFSET.  Copies 64B per lane.
static __device__ __forceinline__ void async_copy64(unsigned lds_addr,
                                                    unsigned voff,
                                                    const void* sbase) {
  asm volatile(
      "global_load_async_to_lds_b128 %0, %1, %2 offset:0\n\t"
      "global_load_async_to_lds_b128 %0, %1, %2 offset:16\n\t"
      "global_load_async_to_lds_b128 %0, %1, %2 offset:32\n\t"
      "global_load_async_to_lds_b128 %0, %1, %2 offset:48"
      :: "v"(lds_addr), "v"(voff), "s"(sbase) : "memory");
}
static __device__ __forceinline__ void wait_async0() {
  asm volatile("s_wait_asynccnt 0x0" ::: "memory");
}

// ---------------------------------------------------------------------------
// Kernel 0: one-shot weight conversion f32 -> bf16 (all four are 32K elems).
// ---------------------------------------------------------------------------
__global__ __launch_bounds__(256) void k_wcvt(
    const float* __restrict__ Wg, const float* __restrict__ Wt,
    const float* __restrict__ Wp, const float* __restrict__ Wo,
    __bf16* __restrict__ Wgb, __bf16* __restrict__ Wtb,
    __bf16* __restrict__ Wpb, __bf16* __restrict__ Wob) {
  int i = blockIdx.x * 256 + threadIdx.x;   // grid covers kCI*kC = 32768
  Wgb[i] = (__bf16)Wg[i];
  Wtb[i] = (__bf16)Wt[i];
  Wpb[i] = (__bf16)Wp[i];
  Wob[i] = (__bf16)Wo[i];
}

// ---------------------------------------------------------------------------
// Kernel 1: fused projections.
//   thT/phT[b][n][ci]  (n-major, K contiguous for score-GEMM fragments)
//   gC[b][ci][n]       (ci-major, so PV B-fragments become contiguous in LDS)
// One WG = 64 spatial rows; wave w owns 16 rows, loops over all 24 o-tiles.
// ---------------------------------------------------------------------------
__global__ __launch_bounds__(128) void k_proj(
    const float* __restrict__ x,
    const __bf16* __restrict__ Wgb, const float* __restrict__ bg,
    const __bf16* __restrict__ Wtb, const float* __restrict__ bt,
    const __bf16* __restrict__ Wpb, const float* __restrict__ bp,
    __bf16* __restrict__ gC, __bf16* __restrict__ thT, __bf16* __restrict__ phT) {
  __shared__ __align__(16) __bf16 xs[kC * 64];   // x tile, [c][n_local], bf16

  int b   = blockIdx.x >> 6;
  int n0  = (blockIdx.x & 63) * 64;
  int tid = threadIdx.x;

  // Cooperative stage of x[b, :, n0:n0+64] into LDS with f32->bf16 convert.
#pragma unroll
  for (int r = 0; r < 2; ++r) {
    int c = tid * 2 + r;
    const float4* src = (const float4*)(x + ((size_t)(b * kC + c)) * kN + n0);
#pragma unroll
    for (int q = 0; q < 16; ++q) {
      float4 f = src[q];
      xs[c * 64 + q * 4 + 0] = (__bf16)f.x;
      xs[c * 64 + q * 4 + 1] = (__bf16)f.y;
      xs[c * 64 + q * 4 + 2] = (__bf16)f.z;
      xs[c * 64 + q * 4 + 3] = (__bf16)f.w;
    }
  }
  __syncthreads();

  int lane = tid & 31, w = tid >> 5;
  int m0 = w * 16;

  // A = x^T tile (16 rows x K=256) held in registers for all 24 o-tiles.
  v16bf a[8];
#pragma unroll
  for (int kk = 0; kk < 8; ++kk) a[kk] = load_aT(xs + kk * 32 * 64 + m0, 64, lane);

  int hl = (lane >> 4) & 1, nl = lane & 15;
  for (int ot = 0; ot < 24; ++ot) {
    int sel = ot >> 3;           // 0->g, 1->theta, 2->phi
    int co  = (ot & 7) * 16;     // column offset inside the selected projection
    const __bf16* Wsel; const float* bsel;
    if (sel == 0)      { Wsel = Wgb; bsel = bg; }
    else if (sel == 1) { Wsel = Wtb; bsel = bt; }
    else               { Wsel = Wpb; bsel = bp; }

    v8f acc = vsplat8(bsel[co + nl]);
#pragma unroll
    for (int kk = 0; kk < 8; ++kk) {
      v16bf bb = load_bT(Wsel + (size_t)co * kC + kk * 32, kC, lane);
      acc = WMMA_BF16(a[kk], bb, acc);
    }
    if (sel == 0) {
      // gC[b][ci][n]: lane-fixed channel, 8 consecutive n per lane.
      __bf16* d = gC + ((size_t)(b * kCI + co + nl)) * kN + n0 + m0 + 8 * hl;
#pragma unroll
      for (int v = 0; v < 8; ++v) d[v] = (__bf16)acc[v];
    } else {
      __bf16* d = (sel == 1 ? thT : phT) + ((size_t)b * kN) * kCI;
#pragma unroll
      for (int v = 0; v < 8; ++v) {
        int row = n0 + m0 + v + 8 * hl;
        d[(size_t)row * kCI + co + nl] = (__bf16)acc[v];
      }
    }
  }
}

// ---------------------------------------------------------------------------
// Kernel 2: flash attention.  y[b][i][ci] = sum_j softmax_j(th_i . ph_j) g[ci,j]
// WG = 4 waves = 64 query rows; j loop in blocks of 32 with online softmax.
// ph slab [j][ci] and g slab [ci][j] are double-buffered in LDS via async
// copies; P=exp(S-m) round-trips through per-wave LDS to re-enter A-layout.
// ---------------------------------------------------------------------------
__global__ __launch_bounds__(128) void k_attn(
    const __bf16* __restrict__ thT, const __bf16* __restrict__ phT,
    const __bf16* __restrict__ gC,  __bf16* __restrict__ yT) {
  __shared__ __align__(16) __bf16 phs[2][32 * kCI];   // [j][ci] slabs (8KB ea)
  __shared__ __align__(16) __bf16 gs [2][kCI * 32];   // [ci][j] slabs (8KB ea)
  __shared__ __align__(16) __bf16 ps [4 * 16 * 32];   // per-wave P tile

  int b    = blockIdx.x >> 6;
  int tid  = threadIdx.x;
  int lane = tid & 31, w = tid >> 5;
  int i0   = (blockIdx.x & 63) * 64 + w * 16;
  int hl   = (lane >> 4) & 1, nl = lane & 15;

  // Query tile A fragments (16 x K=128), kept resident for the whole j loop.
  v16bf ta[4];
#pragma unroll
  for (int kk = 0; kk < 4; ++kk)
    ta[kk] = load_a(thT + ((size_t)(b * kN + i0)) * kCI + kk * 32, kCI, lane);

  v8f acc[8];                       // y accumulator: 16 rows x 128 ci (f32)
#pragma unroll
  for (int t = 0; t < 8; ++t) acc[t] = vsplat8(0.0f);
  float mrow[8], lrow[8];
#pragma unroll
  for (int v = 0; v < 8; ++v) { mrow[v] = -3.0e38f; lrow[v] = 0.0f; }

  // Async prefetch of j-block jb into buffer jb&1.
  auto issue_slab = [&](int jb) {
    int j0 = jb * 32;
    int bi = jb & 1;
    // phT slab: 8KB fully contiguous; 64B per thread.
    const void* pbase = (const void*)(phT + ((size_t)(b * kN + j0)) * kCI);
    unsigned    plds  = (unsigned)(size_t)(&phs[bi][0]) + tid * 64;
    async_copy64(plds, (unsigned)(tid * 64), pbase);
    // gC slab: row per thread (ci = tid), 64B contiguous per row.
    const void* gbase = (const void*)(gC + ((size_t)b * kCI) * kN + j0);
    unsigned    glds  = (unsigned)(size_t)(&gs[bi][0]) + tid * 64;
    async_copy64(glds, (unsigned)(tid * (kN * 2)), gbase);
  };

  issue_slab(0);

  for (int jb = 0; jb < 128; ++jb) {
    wait_async0();          // our slab for jb is in LDS
    __syncthreads();        // everyone's slab portions landed; jb-1 reads done
    if (jb + 1 < 128) issue_slab(jb + 1);

    const __bf16* php = &phs[jb & 1][0];
    const __bf16* gp  = &gs [jb & 1][0];

    // Scores S = th_i^T . ph_j  (16 x 32, two 16x16 C tiles, K=128)
    v8f s0 = vsplat8(0.0f), s1 = vsplat8(0.0f);
#pragma unroll
    for (int kk = 0; kk < 4; ++kk) {
      s0 = WMMA_BF16(ta[kk], load_bT(php + kk * 32,            kCI, lane), s0);
      s1 = WMMA_BF16(ta[kk], load_bT(php + 16 * kCI + kk * 32, kCI, lane), s1);
    }

    // Online softmax (reference applies no 1/sqrt(d) scaling).
    float sc[8]; v8f p0, p1;
#pragma unroll
    for (int v = 0; v < 8; ++v) {
      float mx = half_max(fmaxf(s0[v], s1[v]));
      float mn = fmaxf(mrow[v], mx);
      sc[v]    = __expf(mrow[v] - mn);
      float e0 = __expf(s0[v] - mn);
      float e1 = __expf(s1[v] - mn);
      lrow[v]  = lrow[v] * sc[v] + half_sum(e0 + e1);
      mrow[v]  = mn;
      p0[v] = e0; p1[v] = e1;
    }
#pragma unroll
    for (int t = 0; t < 8; ++t)
#pragma unroll
      for (int v = 0; v < 8; ++v) acc[t][v] *= sc[v];

    // Stage P (bf16) in this wave's private LDS region, reload as A fragment.
    __bf16* pw = ps + w * 16 * 32;
#pragma unroll
    for (int v = 0; v < 8; ++v) {
      int row = v + 8 * hl;
      pw[row * 32 + nl]      = (__bf16)p0[v];
      pw[row * 32 + 16 + nl] = (__bf16)p1[v];
    }
    v16bf pa = load_a(pw, 32, lane);

    // acc += P (16x32) x g^T (32 x 128); gp is [ci][j] so fragments are
    // two contiguous 8-bf16 runs per lane (vectorized ds loads).
#pragma unroll
    for (int ct = 0; ct < 8; ++ct)
      acc[ct] = WMMA_BF16(pa, load_bT(gp + ct * 16 * 32, 32, lane), acc[ct]);
  }

  // Epilogue: y = acc / l, store bf16 [b][n][ci]
#pragma unroll
  for (int ct = 0; ct < 8; ++ct)
#pragma unroll
    for (int v = 0; v < 8; ++v) {
      int row = v + 8 * hl;
      float val = acc[ct][v] / lrow[v];
      yT[((size_t)(b * kN + i0 + row)) * kCI + ct * 16 + nl] = (__bf16)val;
    }
}

// ---------------------------------------------------------------------------
// Kernel 3: output conv + residual.  out[b][o][n] = Wo[o,:] . y[:,n] + bo + x
// ---------------------------------------------------------------------------
__global__ __launch_bounds__(128) void k_out(
    const __bf16* __restrict__ yT, const __bf16* __restrict__ Wob,
    const float* __restrict__ bo,  const float* __restrict__ x,
    float* __restrict__ out) {
  int b    = blockIdx.x >> 6;
  int lane = threadIdx.x & 31, w = threadIdx.x >> 5;
  int n0   = (blockIdx.x & 63) * 64 + w * 16;
  int hl   = (lane >> 4) & 1, nl = lane & 15;

  v16bf a[4];
#pragma unroll
  for (int kk = 0; kk < 4; ++kk)
    a[kk] = load_a(yT + ((size_t)(b * kN + n0)) * kCI + kk * 32, kCI, lane);

  for (int ot = 0; ot < 16; ++ot) {
    int o0 = ot * 16;
    v8f acc = vsplat8(bo[o0 + nl]);
#pragma unroll
    for (int kk = 0; kk < 4; ++kk)
      acc = WMMA_BF16(a[kk], load_bT(Wob + (size_t)o0 * kCI + kk * 32, kCI, lane), acc);
#pragma unroll
    for (int v = 0; v < 8; ++v) {
      size_t off = ((size_t)(b * kC + o0 + nl)) * kN + n0 + v + 8 * hl;
      out[off] = acc[v] + x[off];
    }
  }
}

// ---------------------------------------------------------------------------
extern "C" void kernel_launch(void* const* d_in, const int* in_sizes, int n_in,
                              void* d_out, int out_size, void* d_ws, size_t ws_size,
                              hipStream_t stream) {
  const float* x  = (const float*)d_in[0];
  const float* Wg = (const float*)d_in[1];
  const float* bg = (const float*)d_in[2];
  const float* Wt = (const float*)d_in[3];
  const float* bt = (const float*)d_in[4];
  const float* Wp = (const float*)d_in[5];
  const float* bp = (const float*)d_in[6];
  const float* Wo = (const float*)d_in[7];
  const float* bo = (const float*)d_in[8];

  // Workspace: gC, thT, phT, yT — each [B][N][CI] bf16 = 4 MB; then 4x 64KB
  // bf16 weight copies.  Total ~16.25 MB.
  char*  ws   = (char*)d_ws;
  size_t seg  = (size_t)kB * kN * kCI * sizeof(__bf16);
  size_t wseg = (size_t)kCI * kC * sizeof(__bf16);
  __bf16* gC  = (__bf16*)(ws);
  __bf16* thT = (__bf16*)(ws + seg);
  __bf16* phT = (__bf16*)(ws + 2 * seg);
  __bf16* yT  = (__bf16*)(ws + 3 * seg);
  __bf16* Wgb = (__bf16*)(ws + 4 * seg);
  __bf16* Wtb = (__bf16*)(ws + 4 * seg + wseg);
  __bf16* Wpb = (__bf16*)(ws + 4 * seg + 2 * wseg);
  __bf16* Wob = (__bf16*)(ws + 4 * seg + 3 * wseg);

  dim3 grid(kB * (kN / 64));   // 256 workgroups
  dim3 blk(128);               // 4 waves (wave32)

  k_wcvt<<<dim3((kCI * kC) / 256), dim3(256), 0, stream>>>(Wg, Wt, Wp, Wo,
                                                           Wgb, Wtb, Wpb, Wob);
  k_proj<<<grid, blk, 0, stream>>>(x, Wgb, bg, Wtb, bt, Wpb, bp, gC, thT, phT);
  k_attn<<<grid, blk, 0, stream>>>(thT, phT, gC, yT);
  k_out <<<grid, blk, 0, stream>>>(yT, Wob, bo, x, (float*)d_out);
}
